// Social_Aggregator_6004364279883
// MI455X (gfx1250) — matrix operand order repaired and checked by reading, not compiled, not run on hardware
//
#include <hip/hip_runtime.h>
#include <hip/hip_bf16.h>
#include <stdint.h>

typedef __attribute__((ext_vector_type(16))) __bf16 v16bf;
typedef __attribute__((ext_vector_type(8)))  float  v8f;

#define WAVES 4
#define CSTR  68    // row stride (ushorts) for 32x64 bf16 tiles (+4 pad, 136B rows)
#define W1STR 132   // W1^T row stride (128 + 4 pad)
#define W2STR 68    // W2^T row stride (64 + 4 pad)

__device__ __forceinline__ unsigned short f32_to_bf16(float f) {
  unsigned int x = __float_as_uint(f);
  x += 0x7FFFu + ((x >> 16) & 1u);          // round-to-nearest-even
  return (unsigned short)(x >> 16);
}
__device__ __forceinline__ float bf16_to_f32(unsigned short u) {
  return __uint_as_float(((unsigned int)u) << 16);
}

union FragBF { unsigned int u[8]; v16bf v; };

// A-matrix 16x32 bf16 fragment (ISA 7.12.2): lanes 0-15 row M=lane, K = kb+0..7
// in v0-3 and kb+16..23 in v4-7; lanes 16-31 same rows, K = kb+8..15 / kb+24..31.
__device__ __forceinline__ v16bf load_a_frag(const unsigned short* buf, int stride,
                                             int row0, int kbase, int lane) {
  const int half = lane >> 4;
  const int mr   = lane & 15;
  const unsigned short* p  = buf + (row0 + mr) * stride + kbase + half * 8;
  const unsigned int*   q0 = (const unsigned int*)p;
  const unsigned int*   q1 = (const unsigned int*)(p + 16);
  FragBF f;
  f.u[0] = q0[0]; f.u[1] = q0[1]; f.u[2] = q0[2]; f.u[3] = q0[3];
  f.u[4] = q1[0]; f.u[5] = q1[1]; f.u[6] = q1[2]; f.u[7] = q1[3];
  return f.v;
}

// B-matrix 32x16 bf16 fragment: lane = column (N), lanes 0-15 hold K=kb+0..7 and
// kb+16..23, lanes 16-31 hold K=kb+8..15 and kb+24..31 (weights stored transposed).
__device__ __forceinline__ v16bf load_b_frag(const unsigned short* bufT, int stride,
                                             int ncol0, int kbase, int lane) {
  const int half = lane >> 4;
  const int nc   = lane & 15;
  const unsigned int* q =
      (const unsigned int*)(bufT + (ncol0 + nc) * stride + kbase + half * 8);
  FragBF f;
#pragma unroll
  for (int j = 0; j < 4; ++j) { f.u[j] = q[j]; f.u[j + 4] = q[j + 8]; }
  return f.v;
}

__device__ __forceinline__ v8f wmma_bf16(v16bf a, v16bf b, v8f c) {
  return __builtin_amdgcn_wmma_f32_16x16x32_bf16(false, a, false, b, (short)0, c,
                                                 false, false);
}

// One wave = one attention group: X=[32 x 128]=cat(c,d_bcast), h1=relu(X@W1+b1),
// h2=relu(h1@W2+b2), logits=h2@W3+b3, w=softmax(logits), out=sum_k w_k*c_k.
__global__ __launch_bounds__(WAVES * 32) void attn_agg_kernel(
    const float* __restrict__ cSrc, const int* __restrict__ cIdx,
    const float* __restrict__ dSrc, const int* __restrict__ dIdx,
    const float* __restrict__ gW1, const float* __restrict__ gB1,
    const float* __restrict__ gW2, const float* __restrict__ gB2,
    const float* __restrict__ gW3, const float* __restrict__ gB3,
    float* __restrict__ out, int numGroups) {
  __shared__ unsigned short sW1t[64 * W1STR];         // W1^T [n][k], bf16
  __shared__ unsigned short sW2t[64 * W2STR];         // W2^T [n][k], bf16
  __shared__ float sB1[64], sB2[64], sW3[64];
  __shared__ float sB3;
  __shared__ unsigned short sC[WAVES][32 * CSTR];     // neighbor rows, bf16
  __shared__ unsigned short sH[WAVES][32 * CSTR];     // hidden rows,   bf16
  __shared__ unsigned short sD[WAVES][64];            // center row,    bf16
  __shared__ float sRed[WAVES][32];                   // logits then weights

  const int tid = threadIdx.x;

  // ---- Stage weights once per block (transposed, bf16) ----
  for (int i = tid; i < 128 * 64; i += blockDim.x) {
    int k = i >> 6, n = i & 63;
    sW1t[n * W1STR + k] = f32_to_bf16(gW1[i]);
  }
  for (int i = tid; i < 64 * 64; i += blockDim.x) {
    int k = i >> 6, n = i & 63;
    sW2t[n * W2STR + k] = f32_to_bf16(gW2[i]);
  }
  if (tid < 64) { sB1[tid] = gB1[tid]; sB2[tid] = gB2[tid]; sW3[tid] = gW3[tid]; }
  if (tid == 0) sB3 = gB3[0];
  __syncthreads();

  const int wave = tid >> 5;
  const int lane = tid & 31;
  const int g    = blockIdx.x * WAVES + wave;
  if (g >= numGroups) return;

  const int half = lane >> 4;
  const int nc   = lane & 15;

  // ---- Gather 32 neighbor rows (one per lane) + center row into LDS ----
  {
    long long srow = cIdx ? (long long)cIdx[(long long)g * 32 + lane]
                          : ((long long)g * 32 + lane);
    const float* src = cSrc + srow * 64;
    unsigned short* crow = &sC[wave][lane * CSTR];
#pragma unroll
    for (int j = 0; j < 64; j += 4) {
      float4 v = *(const float4*)(src + j);
      *(unsigned int*)(crow + j) =
          (unsigned int)f32_to_bf16(v.x) | ((unsigned int)f32_to_bf16(v.y) << 16);
      *(unsigned int*)(crow + j + 2) =
          (unsigned int)f32_to_bf16(v.z) | ((unsigned int)f32_to_bf16(v.w) << 16);
    }
    const float* dsrc = dSrc + (long long)dIdx[g] * 64;
    float2 dv = *(const float2*)(dsrc + lane * 2);
    *(unsigned int*)(&sD[wave][lane * 2]) =
        (unsigned int)f32_to_bf16(dv.x) | ((unsigned int)f32_to_bf16(dv.y) << 16);
  }
  __asm__ volatile("s_wait_dscnt 0" ::: "memory");

  v8f zero;
#pragma unroll
  for (int i = 0; i < 8; ++i) zero[i] = 0.0f;

  // ---- Matmul 1: [32x128] @ [128x64], K-chunks 0-1 from c, 2-3 from d ----
  v8f acc1[2][4];
#pragma unroll
  for (int m = 0; m < 2; ++m)
#pragma unroll
    for (int n = 0; n < 4; ++n) acc1[m][n] = zero;

#pragma unroll
  for (int kk = 0; kk < 4; ++kk) {
    v16bf a0, a1;
    if (kk < 2) {
      a0 = load_a_frag(&sC[wave][0], CSTR, 0,  kk * 32, lane);
      a1 = load_a_frag(&sC[wave][0], CSTR, 16, kk * 32, lane);
    } else {  // broadcast center row (stride 0 -> same data in all M rows)
      a0 = load_a_frag(&sD[wave][0], 0, 0, (kk - 2) * 32, lane);
      a1 = a0;
    }
#pragma unroll
    for (int n = 0; n < 4; ++n) {
      v16bf b = load_b_frag(sW1t, W1STR, n * 16, kk * 32, lane);
      acc1[0][n] = wmma_bf16(a0, b, acc1[0][n]);
      acc1[1][n] = wmma_bf16(a1, b, acc1[1][n]);
    }
  }

  // relu(acc + b1) -> sH as bf16 (C/D layout: reg r, lane -> row 16m+8*half+r, col 16n+nc)
#pragma unroll
  for (int m = 0; m < 2; ++m)
#pragma unroll
    for (int n = 0; n < 4; ++n) {
      int col = n * 16 + nc;
      float bias = sB1[col];
#pragma unroll
      for (int r = 0; r < 8; ++r) {
        float hv = fmaxf(acc1[m][n][r] + bias, 0.0f);
        int row = m * 16 + half * 8 + r;
        sH[wave][row * CSTR + col] = f32_to_bf16(hv);
      }
    }
  __asm__ volatile("s_wait_dscnt 0" ::: "memory");

  // ---- Matmul 2: [32x64] @ [64x64] ----
  v8f acc2[2][4];
#pragma unroll
  for (int m = 0; m < 2; ++m)
#pragma unroll
    for (int n = 0; n < 4; ++n) acc2[m][n] = zero;

#pragma unroll
  for (int kk = 0; kk < 2; ++kk) {
    v16bf a0 = load_a_frag(&sH[wave][0], CSTR, 0,  kk * 32, lane);
    v16bf a1 = load_a_frag(&sH[wave][0], CSTR, 16, kk * 32, lane);
#pragma unroll
    for (int n = 0; n < 4; ++n) {
      v16bf b = load_b_frag(sW2t, W2STR, n * 16, kk * 32, lane);
      acc2[0][n] = wmma_bf16(a0, b, acc2[0][n]);
      acc2[1][n] = wmma_bf16(a1, b, acc2[1][n]);
    }
  }

  // ---- logits = relu(acc2 + b2) @ W3 : per-lane partials + half-wave reduction ----
  float part[2][8];
#pragma unroll
  for (int m = 0; m < 2; ++m)
#pragma unroll
    for (int r = 0; r < 8; ++r) part[m][r] = 0.0f;

#pragma unroll
  for (int m = 0; m < 2; ++m)
#pragma unroll
    for (int n = 0; n < 4; ++n) {
      int col = n * 16 + nc;
      float b2v = sB2[col], w3v = sW3[col];
#pragma unroll
      for (int r = 0; r < 8; ++r)
        part[m][r] += fmaxf(acc2[m][n][r] + b2v, 0.0f) * w3v;
    }
#pragma unroll
  for (int m = 0; m < 2; ++m)
#pragma unroll
    for (int r = 0; r < 8; ++r) {
      float s = part[m][r];
      s += __shfl_xor(s, 1);
      s += __shfl_xor(s, 2);
      s += __shfl_xor(s, 4);
      s += __shfl_xor(s, 8);          // stays inside each 16-lane half (wave32)
      if (nc == 0) sRed[wave][m * 16 + half * 8 + r] = s + sB3;
    }
  __asm__ volatile("s_wait_dscnt 0" ::: "memory");

  // ---- softmax over the 32 logits (row = lane) ----
  float lg = sRed[wave][lane];
  float mx = lg;
#pragma unroll
  for (int off = 16; off >= 1; off >>= 1) mx = fmaxf(mx, __shfl_xor(mx, off));
  float e = __expf(lg - mx);
  float se = e;
#pragma unroll
  for (int off = 16; off >= 1; off >>= 1) se += __shfl_xor(se, off);
  sRed[wave][lane] = e / se;
  __asm__ volatile("s_wait_dscnt 0" ::: "memory");

  // ---- out[col] = sum_k w[k] * c[k][col]; each lane covers 2 columns ----
#pragma unroll
  for (int cpart = 0; cpart < 2; ++cpart) {
    int col = lane + cpart * 32;
    float o = 0.0f;
#pragma unroll 8
    for (int k = 0; k < 32; ++k)
      o += sRed[wave][k] * bf16_to_f32(sC[wave][k * CSTR + col]);
    out[(long long)g * 64 + col] = o;
  }
}

extern "C" void kernel_launch(void* const* d_in, const int* in_sizes, int n_in,
                              void* d_out, int out_size, void* d_ws, size_t ws_size,
                              hipStream_t stream) {
  const int*   nodes     = (const int*)d_in[0];
  const int*   to_neighs = (const int*)d_in[1];
  const int*   to_n1     = (const int*)d_in[2];
  const float* u2e       = (const float*)d_in[3];
  const float* W1        = (const float*)d_in[4];
  const float* b1        = (const float*)d_in[5];
  const float* W2        = (const float*)d_in[6];
  const float* b2        = (const float*)d_in[7];
  const float* W3        = (const float*)d_in[8];
  const float* b3        = (const float*)d_in[9];

  const int N       = in_sizes[0];                    // 2048
  const int K1      = (N > 0) ? (in_sizes[1] / N) : 32;  // 32
  const int groups1 = N * K1;                         // 65536
  float* e_u = (float*)d_ws;                          // [N*K1, 64] fp32 (16 MB)

  dim3 block(WAVES * 32);
  // Level 1: c = u2e[to_neighs_1], d = u2e[to_neighs] -> e_u
  attn_agg_kernel<<<dim3((groups1 + WAVES - 1) / WAVES), block, 0, stream>>>(
      u2e, to_n1, u2e, to_neighs, W1, b1, W2, b2, W3, b3, e_u, groups1);
  // Level 2: c = e_u (contiguous rows), d = u2e[nodes] -> d_out
  attn_agg_kernel<<<dim3((N + WAVES - 1) / WAVES), block, 0, stream>>>(
      e_u, nullptr, u2e, nodes, W1, b1, W2, b2, W3, b3, (float*)d_out, N);
}
